// GCN_90288802497036
// MI455X (gfx1250) — compile-verified, matrix operand order checked by reading, not run on hardware
//
#include <hip/hip_runtime.h>
#include <hip/hip_bf16.h>

typedef __attribute__((ext_vector_type(16))) __bf16 v16bf;
typedef __attribute__((ext_vector_type(8)))  float  v8f;
typedef __attribute__((ext_vector_type(4)))  float  v4f;

#define N_NODES 100000
#define E_EDGES 3200000
#define IN_F    512
#define HID     16
#define OUT_F   7
#define N_TILES (N_NODES / 16)   // 6250
#define K_CHUNKS (IN_F / 32)     // 16

// Native fp32 global atomic add (relaxed, device scope) -> global_atomic_add_f32
__device__ __forceinline__ void atomic_add_f32(float* p, float v) {
    __hip_atomic_fetch_add(p, v, __ATOMIC_RELAXED, __HIP_MEMORY_SCOPE_AGENT);
}

// ---------------------------------------------------------------------------
// Pack W1 [512,16] into the exact bf16 B-matrix VGPR layout for
// v_wmma_f32_16x16x32_bf16, split into hi/lo bf16 parts for precision.
// Layout per chunk c (K = c*32..c*32+31), lane L (n = L%16, g = L/16),
// vgpr v, half h:  k = (v<4) ? c*32 + g*8 + 2v + h : c*32 + 16 + g*8 + 2(v-4) + h
// ---------------------------------------------------------------------------
__global__ void GCN_pack_w1(const float* __restrict__ W1,
                            __bf16* __restrict__ Bh, __bf16* __restrict__ Bl) {
    int id = blockIdx.x * blockDim.x + threadIdx.x;   // c*256 + lane*8 + v
    if (id >= K_CHUNKS * 32 * 8) return;
    int v    = id & 7;
    int lane = (id >> 3) & 31;
    int c    = id >> 8;
    int n    = lane & 15;
    int g    = lane >> 4;
    int kb   = c * 32;
#pragma unroll
    for (int h = 0; h < 2; ++h) {
        int k = (v < 4) ? (kb + g * 8 + 2 * v + h)
                        : (kb + 16 + g * 8 + 2 * (v - 4) + h);
        float w   = W1[k * HID + n];
        __bf16 hi = (__bf16)w;
        __bf16 lo = (__bf16)(w - (float)hi);
        int idx = (c * 32 + lane) * 16 + v * 2 + h;
        Bh[idx] = hi;
        Bl[idx] = lo;
    }
}

// ---------------------------------------------------------------------------
// GEMM1: h0[N,16] = x[N,512] @ W1[512,16] via bf16 WMMA with hi/lo split
// (3 WMMAs per K-chunk: ah*bh + al*bh + ah*bl, f32 accumulate).
// One wave per 16-row tile; 8 waves per block. HBM-bound on the 205 MB x read.
// ---------------------------------------------------------------------------
__global__ void __launch_bounds__(256)
GCN_gemm1_wmma(const float* __restrict__ x,
               const __bf16* __restrict__ Bh, const __bf16* __restrict__ Bl,
               float* __restrict__ h0) {
    const int wave = threadIdx.x >> 5;
    const int lane = threadIdx.x & 31;
    const int tile = blockIdx.x * 8 + wave;
    if (tile >= N_TILES) return;               // wave-uniform: EXEC stays all-1s

    const int n = lane & 15;                   // A row m / C col n
    const int g = lane >> 4;
    const float* xrow = x + (size_t)(tile * 16 + n) * IN_F;

    v8f acc = {};
#pragma unroll 4
    for (int c = 0; c < K_CHUNKS; ++c) {
        const int kb = c * 32;
        // 16 contiguous-in-K f32 values of this lane's A row (two 32B spans);
        // lanes m and m+16 together cover one full 128B span of row m.
        float f[16];
        *(v4f*)(f + 0)  = *(const v4f*)(xrow + kb + g * 8);
        *(v4f*)(f + 4)  = *(const v4f*)(xrow + kb + g * 8 + 4);
        *(v4f*)(f + 8)  = *(const v4f*)(xrow + kb + 16 + g * 8);
        *(v4f*)(f + 12) = *(const v4f*)(xrow + kb + 16 + g * 8 + 4);
        if (c + 1 < K_CHUNKS)
            __builtin_prefetch(xrow + kb + 32 + g * 8, 0, 3);  // WGP-scope

        v16bf ah, al;
#pragma unroll
        for (int i = 0; i < 16; ++i) {
            __bf16 hi = (__bf16)f[i];
            ah[i] = hi;
            al[i] = (__bf16)(f[i] - (float)hi);
        }
        v16bf bh = *(const v16bf*)(Bh + (c * 32 + lane) * 16);
        v16bf bl = *(const v16bf*)(Bl + (c * 32 + lane) * 16);

        acc = __builtin_amdgcn_wmma_f32_16x16x32_bf16(false, ah, false, bh,
                                                      (short)0, acc, false, false);
        acc = __builtin_amdgcn_wmma_f32_16x16x32_bf16(false, al, false, bh,
                                                      (short)0, acc, false, false);
        acc = __builtin_amdgcn_wmma_f32_16x16x32_bf16(false, ah, false, bl,
                                                      (short)0, acc, false, false);
    }
    // C/D layout: vgpr r, lane L -> (m = r + (L>=16)*8, n = L%16)
#pragma unroll
    for (int r = 0; r < 8; ++r)
        h0[(size_t)(tile * 16 + r + g * 8) * HID + n] = acc[r];
}

// ---------------------------------------------------------------------------
// SPMM, d=16: agg[dst,j] += w[e] * h[src,j]; one thread per (edge, feature).
// 16 consecutive lanes hit one 64B row: coalesced gather + coalesced atomics.
// Accumulator buffers are L2-resident (6.4 MB << 192 MB).
// ---------------------------------------------------------------------------
__global__ void GCN_spmm16(const int* __restrict__ ei, const float* __restrict__ ew,
                           const float* __restrict__ h, float* __restrict__ agg) {
    long long t = (long long)blockIdx.x * blockDim.x + threadIdx.x;
    if (t >= (long long)E_EDGES * 16) return;
    int e = (int)(t >> 4);
    int j = (int)(t & 15);
    int src = ei[e];
    int dst = ei[E_EDGES + e];
    float v = ew[e] * h[(size_t)src * HID + j];
    atomic_add_f32(&agg[(size_t)dst * HID + j], v);
}

// SPMM, d=7 (8 lanes per edge, lane 7 idles)
__global__ void GCN_spmm7(const int* __restrict__ ei, const float* __restrict__ ew,
                          const float* __restrict__ z, float* __restrict__ agg) {
    long long t = (long long)blockIdx.x * blockDim.x + threadIdx.x;
    if (t >= (long long)E_EDGES * 8) return;
    int e = (int)(t >> 3);
    int j = (int)(t & 7);
    if (j >= OUT_F) return;
    int src = ei[e];
    int dst = ei[E_EDGES + e];
    float v = ew[e] * z[(size_t)src * OUT_F + j];
    atomic_add_f32(&agg[(size_t)dst * OUT_F + j], v);
}

// ---------------------------------------------------------------------------
// Fused bias + ReLU + GEMM2:  z[n,0:7] = relu(agg1[n,:] + b1) @ W2
// W2 (112 f32) and b1 staged in LDS. One node per thread.
// ---------------------------------------------------------------------------
__global__ void __launch_bounds__(256)
GCN_relu_gemm2(const float* __restrict__ agg1, const float* __restrict__ b1,
               const float* __restrict__ W2, float* __restrict__ z) {
    __shared__ float sW2[HID * OUT_F];
    __shared__ float sb1[HID];
    if (threadIdx.x < HID * OUT_F) sW2[threadIdx.x] = W2[threadIdx.x];
    if (threadIdx.x < HID)         sb1[threadIdx.x] = b1[threadIdx.x];
    __syncthreads();

    int node = blockIdx.x * blockDim.x + threadIdx.x;
    if (node >= N_NODES) return;

    float hv[HID];
#pragma unroll
    for (int i = 0; i < HID; ++i)
        hv[i] = fmaxf(agg1[(size_t)node * HID + i] + sb1[i], 0.0f);
#pragma unroll
    for (int j = 0; j < OUT_F; ++j) {
        float s = 0.0f;
#pragma unroll
        for (int i = 0; i < HID; ++i)
            s = fmaf(hv[i], sW2[i * OUT_F + j], s);
        z[(size_t)node * OUT_F + j] = s;
    }
}

// Bias + log_softmax over 7 classes. One node per thread.
__global__ void __launch_bounds__(256)
GCN_logsoftmax(const float* __restrict__ agg2, const float* __restrict__ b2,
               float* __restrict__ out) {
    __shared__ float sb2[OUT_F];
    if (threadIdx.x < OUT_F) sb2[threadIdx.x] = b2[threadIdx.x];
    __syncthreads();

    int node = blockIdx.x * blockDim.x + threadIdx.x;
    if (node >= N_NODES) return;

    float v[OUT_F];
    float m = -3.402823466e38f;
#pragma unroll
    for (int j = 0; j < OUT_F; ++j) {
        v[j] = agg2[(size_t)node * OUT_F + j] + sb2[j];
        m = fmaxf(m, v[j]);
    }
    float s = 0.0f;
#pragma unroll
    for (int j = 0; j < OUT_F; ++j) s += expf(v[j] - m);
    float lse = logf(s);
#pragma unroll
    for (int j = 0; j < OUT_F; ++j)
        out[(size_t)node * OUT_F + j] = v[j] - m - lse;
}

__global__ void GCN_zero(float* __restrict__ p, long long n) {
    long long i = (long long)blockIdx.x * blockDim.x + threadIdx.x;
    if (i < n) p[i] = 0.0f;
}

// ---------------------------------------------------------------------------
// launch
// ---------------------------------------------------------------------------
extern "C" void kernel_launch(void* const* d_in, const int* in_sizes, int n_in,
                              void* d_out, int out_size, void* d_ws, size_t ws_size,
                              hipStream_t stream) {
    const float* x  = (const float*)d_in[0];
    const int*   ei = (const int*)d_in[1];      // [2, E]
    const float* ew = (const float*)d_in[2];
    const float* W1 = (const float*)d_in[3];
    const float* b1 = (const float*)d_in[4];
    const float* W2 = (const float*)d_in[5];
    const float* b2 = (const float*)d_in[6];
    float* out = (float*)d_out;

    char* ws = (char*)d_ws;
    size_t off = 0;
    auto alloc = [&](size_t bytes) { char* p = ws + off; off = (off + bytes + 255) & ~(size_t)255; return p; };
    float*  h0   = (float*)alloc((size_t)N_NODES * HID * 4);    // x @ W1
    float*  agg1 = (float*)alloc((size_t)N_NODES * HID * 4);    // spmm layer 1
    float*  z    = (float*)alloc((size_t)N_NODES * OUT_F * 4);  // h @ W2
    float*  agg2 = (float*)alloc((size_t)N_NODES * OUT_F * 4);  // spmm layer 2
    __bf16* Bh   = (__bf16*)alloc((size_t)K_CHUNKS * 32 * 16 * 2);
    __bf16* Bl   = (__bf16*)alloc((size_t)K_CHUNKS * 32 * 16 * 2);

    // zero accumulators (harness poisons ws; atomics need clean zeros)
    {
        long long n1 = (long long)N_NODES * HID;
        long long n2 = (long long)N_NODES * OUT_F;
        GCN_zero<<<(int)((n1 + 255) / 256), 256, 0, stream>>>(agg1, n1);
        GCN_zero<<<(int)((n2 + 255) / 256), 256, 0, stream>>>(agg2, n2);
    }

    // pack W1 into WMMA B layout (hi/lo bf16 split)
    GCN_pack_w1<<<K_CHUNKS, 256, 0, stream>>>(W1, Bh, Bl);

    // layer 1 GEMM: one wave per 16-row tile, 8 waves per block
    GCN_gemm1_wmma<<<(N_TILES + 7) / 8, 256, 0, stream>>>(x, Bh, Bl, h0);

    // layer 1 SPMM
    {
        long long t = (long long)E_EDGES * 16;
        GCN_spmm16<<<(int)((t + 255) / 256), 256, 0, stream>>>(ei, ew, h0, agg1);
    }

    // bias + relu + layer 2 GEMM
    GCN_relu_gemm2<<<(N_NODES + 255) / 256, 256, 0, stream>>>(agg1, b1, W2, z);

    // layer 2 SPMM
    {
        long long t = (long long)E_EDGES * 8;
        GCN_spmm7<<<(int)((t + 255) / 256), 256, 0, stream>>>(ei, ew, z, agg2);
    }

    // bias + log_softmax
    GCN_logsoftmax<<<(N_NODES + 255) / 256, 256, 0, stream>>>(agg2, b2, out);
}
